// DFConv2d_32263794328380
// MI455X (gfx1250) — compile-verified, matrix operand order checked by reading, not run on hardware
//
#include <hip/hip_runtime.h>
#include <hip/hip_bf16.h>

// ---------------------------------------------------------------------------
// Deformable Conv v2 for MI455X (gfx1250, wave32, WMMA).
//   B=4, C=256, H=W=64, O=256, K=3, PAD=1, STRIDE=1, DIL=1
// Stage 1: offset conv  om = conv3x3(x, w_off)+b_off  (sigmoid on mask chans)
// Stage 2: weight repack fp32 (O,C,3,3) -> bf16 in WMMA fragment layout
//          Wf[ks=tap*8+cs][mfrag][lane][16 halves]  (A read straight from L2)
// Stage 3: fused bilinear-gather + implicit GEMM (M=256, K=2304, N=16384)
//          v_wmma_f32_16x16x32_bf16, f32 accumulate, double-buffered B tile.
// ---------------------------------------------------------------------------

#define CIN   256
#define OUTC  256
#define BATCH 4
#define HDIM  64
#define WDIM  64
#define HW    4096           // 64*64
#define NPIX  (BATCH * HW)   // 16384
#define K2    9
#define KSTEPS 72            // 9 taps * 8 channel-steps of 32

typedef __attribute__((ext_vector_type(16))) __bf16 v16bf;
typedef __attribute__((ext_vector_type(8)))  float  v8f;

// Two 16B chunks -> one 16x32 bf16 WMMA fragment (32 bytes / lane).
union Frag32B {
    struct { uint4 lo; uint4 hi; } q;
    v16bf v;
    __bf16 h[16];
};

// ---------------------------------------------------------------------------
// Kernel 1: offset conv. One thread per (b, oc, h, w). 442368 threads.
// Channels >= 18 get sigmoid applied (they are the modulation mask).
// ---------------------------------------------------------------------------
__global__ __launch_bounds__(256) void dfc_offset_conv(
    const float* __restrict__ x, const float* __restrict__ w_off,
    const float* __restrict__ b_off, float* __restrict__ om)
{
    const int id  = blockIdx.x * 256 + threadIdx.x;      // < 4*27*4096
    const int b   = id / (27 * HW);
    const int rem = id - b * (27 * HW);
    const int oc  = rem >> 12;
    const int hw  = rem & (HW - 1);
    const int h   = hw >> 6;
    const int w   = hw & 63;

    const float* xb = x + b * (CIN * HW);
    const float* wr = w_off + oc * (CIN * K2);
    float acc = b_off[oc];

    for (int c = 0; c < CIN; ++c) {
        const float* xc = xb + c * HW;
        const float* wc = wr + c * K2;
#pragma unroll
        for (int ki = 0; ki < 3; ++ki) {
            const int yy = h - 1 + ki;
            if ((unsigned)yy < (unsigned)HDIM) {
                const float* xr = xc + yy * WDIM;
#pragma unroll
                for (int kj = 0; kj < 3; ++kj) {
                    const int xx = w - 1 + kj;
                    if ((unsigned)xx < (unsigned)WDIM)
                        acc = fmaf(xr[xx], wc[ki * 3 + kj], acc);
                }
            }
        }
    }
    if (oc >= 18) acc = 1.0f / (1.0f + __expf(-acc));
    om[id] = acc;
}

// ---------------------------------------------------------------------------
// Kernel 2: repack w_conv (O, C, 3, 3) fp32 -> bf16 directly in the WMMA
// 16x32 A-fragment per-lane layout:
//   Wf[(tap*8+cs)*16 + mfrag][lane][h]  with
//   m  = mfrag*16 + (lane&15),  kh = (lane>>4)*8,
//   kk = h<8 ? kh+h : kh+16+(h-8),   c = cs*32 + kk.
// GEMM then reads A fragments as two coalesced global_load_b128 per lane.
// ---------------------------------------------------------------------------
__global__ __launch_bounds__(256) void dfc_repack_w(
    const float* __restrict__ w_conv, __bf16* __restrict__ Wf)
{
    const int id   = blockIdx.x * 256 + threadIdx.x;     // < 9*8*16*32*16
    const int h    = id & 15;
    const int lane = (id >> 4) & 31;
    const int mf   = (id >> 9) & 15;
    const int cs   = (id >> 13) & 7;
    const int tap  = id >> 16;

    const int m  = mf * 16 + (lane & 15);
    const int kh = (lane >> 4) * 8;
    const int kk = (h < 8) ? (kh + h) : (kh + 16 + (h - 8));
    const int c  = cs * 32 + kk;
    Wf[id] = (__bf16)w_conv[m * (CIN * K2) + c * K2 + tap];
}

// ---------------------------------------------------------------------------
// Per-(pixel, tap) bilinear sampling state: 4 corner weights (validity and
// modulation mask folded in) + 4 clamped gather offsets. Reused for all C.
// ---------------------------------------------------------------------------
struct Coords { float w00, w01, w10, w11; int a00, a01, a10, a11; };

__device__ __forceinline__ Coords make_coords(const float* __restrict__ omb,
                                              int ph, int pw, int phw, int tap)
{
    const float offy = omb[(2 * tap) * HW + phw];
    const float offx = omb[(2 * tap + 1) * HW + phw];
    const float mval = omb[(18 + tap) * HW + phw];        // pre-sigmoided
    const float yy = (float)(ph - 1 + tap / 3) + offy;
    const float xx = (float)(pw - 1 + tap % 3) + offx;
    const float y0f = floorf(yy), x0f = floorf(xx);
    const float ly = yy - y0f, lx = xx - x0f;
    const float hy = 1.0f - ly, hx = 1.0f - lx;
    const int y0 = (int)y0f, x0 = (int)x0f;
    const bool vy0 = (unsigned)y0 < (unsigned)HDIM;
    const bool vy1 = (unsigned)(y0 + 1) < (unsigned)HDIM;
    const bool vx0 = (unsigned)x0 < (unsigned)WDIM;
    const bool vx1 = (unsigned)(x0 + 1) < (unsigned)WDIM;
    Coords cd;
    cd.w00 = hy * hx * mval * ((vy0 && vx0) ? 1.f : 0.f);
    cd.w01 = hy * lx * mval * ((vy0 && vx1) ? 1.f : 0.f);
    cd.w10 = ly * hx * mval * ((vy1 && vx0) ? 1.f : 0.f);
    cd.w11 = ly * lx * mval * ((vy1 && vx1) ? 1.f : 0.f);
    const int y0c = min(max(y0, 0), HDIM - 1);
    const int y1c = min(max(y0 + 1, 0), HDIM - 1);
    const int x0c = min(max(x0, 0), WDIM - 1);
    const int x1c = min(max(x0 + 1, 0), WDIM - 1);
    cd.a00 = y0c * WDIM + x0c; cd.a01 = y0c * WDIM + x1c;
    cd.a10 = y1c * WDIM + x0c; cd.a11 = y1c * WDIM + x1c;
    return cd;
}

// Fill one 32(K) x 128(N) B tile. Each thread gathers 16 *contiguous*
// channels for one pixel, packs to bf16, writes as 2x ds_store_b128.
__device__ __forceinline__ void fill_B(__bf16* __restrict__ buf,
                                       const float* __restrict__ xb,
                                       const Coords& cd, int cbase,
                                       int frow, int fgrp)
{
    Frag32B pk;
#pragma unroll
    for (int i = 0; i < 16; ++i) {
        const int cl = fgrp * 16 + i;                     // k index 0..31
        const float* xc = xb + (cbase + cl) * HW;
        const float v = cd.w00 * xc[cd.a00] + cd.w01 * xc[cd.a01] +
                        cd.w10 * xc[cd.a10] + cd.w11 * xc[cd.a11];
        pk.h[i] = (__bf16)v;
    }
    // (frow*40 + fgrp*16) halves = frow*80 + fgrp*32 bytes -> 16B aligned.
    uint4* d = (uint4*)&buf[frow * 40 + fgrp * 16];
    d[0] = pk.q.lo;
    d[1] = pk.q.hi;
}

// ---------------------------------------------------------------------------
// Kernel 3: fused deformable-gather + WMMA GEMM.
// Block = 256 threads = 8 waves; tile 128(M) x 128(N); wave = 64x32 = 4x2
// accumulators. K loop: 72 steps of 32. B tile double-buffered in LDS so the
// next step's gather overlaps the current step's WMMAs (1 barrier / step).
// A fragments stream straight from L2 in fragment layout (no LDS).
// ---------------------------------------------------------------------------
__global__ __launch_bounds__(256) void dfc_gemm_wmma(
    const float* __restrict__ x, const float* __restrict__ om,
    const __bf16* __restrict__ Wf, float* __restrict__ out)
{
    // Padded row stride of 40 halves (80B) spreads LDS banks.
    __shared__ __bf16 ldsB[2][128 * 40];   // B tiles: [n][k], double buffered

    const int t    = threadIdx.x;
    const int lane = t & 31;
    const int wv   = t >> 5;                 // wave id 0..7
    const int wm   = (wv & 1) * 64;          // wave M offset in tile
    const int wn   = (wv >> 1) * 32;         // wave N offset in tile
    const int m0   = blockIdx.y * 128;       // tile M origin (output channel)
    const int n0   = blockIdx.x * 128;       // tile N origin (pixel)

    // Fill-role identities: thread owns one pixel row and 16 contiguous k's.
    const int frow = t & 127;                // pixel-in-tile
    const int fgrp = t >> 7;                 // 0/1: low/high 16 of the 32 k's

    // ---- per-thread pixel decode (constant across whole kernel) ----
    const int n_g = n0 + frow;
    const int pb  = n_g >> 12;               // batch
    const int phw = n_g & (HW - 1);
    const int ph  = phw >> 6;
    const int pw  = phw & 63;
    const float* xb  = x  + pb * (CIN * HW);
    const float* omb = om + pb * (27 * HW);

    const v8f vzero = {0.f, 0.f, 0.f, 0.f, 0.f, 0.f, 0.f, 0.f};
    v8f acc[4][2];
#pragma unroll
    for (int i = 0; i < 4; ++i)
#pragma unroll
        for (int j = 0; j < 2; ++j) acc[i][j] = vzero;

    // Base of this wave's A fragments within the fragment-layout weights.
    const int mfbase = (m0 + wm) >> 4;       // first of 4 consecutive mfrags

    // ---- prologue: fill buffer 0 for ks = 0 ----
    Coords cd = make_coords(omb, ph, pw, phw, 0);
    fill_B(ldsB[0], xb, cd, 0, frow, fgrp);
    __syncthreads();

    for (int ks = 0; ks < KSTEPS; ++ks) {
        const int cur = ks & 1;

        // ---- overlap: gather next B tile into the other buffer ----
        if (ks + 1 < KSTEPS) {
            const int ksn = ks + 1;
            const int tapn = ksn >> 3;
            const int csn  = ksn & 7;
            if (csn == 0) cd = make_coords(omb, ph, pw, phw, tapn);
            fill_B(ldsB[cur ^ 1], xb, cd, csn * 32, frow, fgrp);
        }

        // ---- A fragments: two coalesced b128 loads per lane, from L2 ----
        const __bf16* Ab = Wf + ((size_t)ks * 16 + mfbase) * 512 + lane * 16;
        if (ks + 1 < KSTEPS)                  // prefetch next step's A block
            __builtin_prefetch(Ab + 16 * 512, 0, 1);
        Frag32B afr[4];
#pragma unroll
        for (int i = 0; i < 4; ++i) {
            const uint4* p = (const uint4*)(Ab + i * 512);
            afr[i].q.lo = p[0];
            afr[i].q.hi = p[1];
        }

        // ---- B fragments from LDS (ISA 16-bit 16x32 per-lane layout):
        //   lanes 0-15:  row = lane,    K halves {0..7, 16..23}
        //   lanes 16-31: row = lane-16, K halves {8..15, 24..31} ----
        const int rsel = lane & 15;
        const int kh   = (lane >> 4) * 8;
        Frag32B bfr[2];
#pragma unroll
        for (int j = 0; j < 2; ++j) {
            const uint4* p =
                (const uint4*)&ldsB[cur][(wn + j * 16 + rsel) * 40 + kh];
            bfr[j].q.lo = p[0];               // halves kh .. kh+7
            bfr[j].q.hi = p[2];               // halves kh+16 .. kh+23
        }

        // Keep every fragment load issued (in flight) before the first WMMA:
        // one L2/LDS latency absorbed per K-step instead of per fragment.
        __builtin_amdgcn_sched_barrier(0);

        // ---- 8 matrix ops: D = A x B + C, f32 accumulate ----
#pragma unroll
        for (int i = 0; i < 4; ++i)
#pragma unroll
            for (int j = 0; j < 2; ++j)
                acc[i][j] = __builtin_amdgcn_wmma_f32_16x16x32_bf16(
                    false, afr[i].v, false, bfr[j].v,
                    (short)0, acc[i][j], false, false);

        __syncthreads();   // next-tile fill complete / current tile consumed
    }

    // ---- epilogue: C/D layout = VGPR r: lanes 0-15 -> M=r, 16-31 -> M=8+r ----
#pragma unroll
    for (int j = 0; j < 2; ++j) {
        const int ncol = n0 + wn + j * 16 + (lane & 15);
        const int bb   = ncol >> 12;
        const int nhw  = ncol & (HW - 1);
        float* op = out + bb * (OUTC * HW) + nhw;
#pragma unroll
        for (int i = 0; i < 4; ++i) {
            const int obase = m0 + wm + i * 16 + ((lane >> 4) << 3);
#pragma unroll
            for (int r = 0; r < 8; ++r)
                op[(obase + r) * HW] = acc[i][j][r];
        }
    }
}

// ---------------------------------------------------------------------------
// Host launcher. Inputs: x, w_off, b_off, w_conv (all fp32). Output fp32.
// Workspace: om (1.77 MB fp32) + Wf (1.18 MB bf16).
// ---------------------------------------------------------------------------
extern "C" void kernel_launch(void* const* d_in, const int* in_sizes, int n_in,
                              void* d_out, int out_size, void* d_ws, size_t ws_size,
                              hipStream_t stream) {
    const float* x      = (const float*)d_in[0];
    const float* w_off  = (const float*)d_in[1];
    const float* b_off  = (const float*)d_in[2];
    const float* w_conv = (const float*)d_in[3];
    float* out = (float*)d_out;

    float*  om = (float*)d_ws;                                   // 4*27*4096 f32
    __bf16* Wf = (__bf16*)((char*)d_ws + (size_t)BATCH * 27 * HW * sizeof(float));

    // Stage 1: offset conv (4*27*4096 outputs, 256 threads/block)
    dfc_offset_conv<<<(BATCH * 27 * HW) / 256, 256, 0, stream>>>(x, w_off, b_off, om);

    // Stage 2: weight repack -> bf16 WMMA-fragment layout
    dfc_repack_w<<<(K2 * CIN * OUTC) / 256, 256, 0, stream>>>(w_conv, Wf);

    // Stage 3: fused gather + WMMA GEMM. Grid: 128 N-tiles x 2 M-tiles.
    dim3 grid(NPIX / 128, OUTC / 128);
    dfc_gemm_wmma<<<grid, 256, 0, stream>>>(x, om, Wf, out);
}
